// FullAttnResTransformerLayer_3539053052690
// MI455X (gfx1250) — compile-verified, matrix operand order checked by reading, not run on hardware
//
#include <hip/hip_runtime.h>
#include <hip/hip_bf16.h>
#include <math.h>

#define LL   6
#define BB   4
#define TT   1024
#define DD   1024
#define HH   16
#define HDIM 64
#define DFF  4096
#define BT   (BB * TT)

#define RMS_EPS 1.1920929e-07f
#define LN_EPS  1e-5f

typedef __attribute__((ext_vector_type(16))) __bf16 v16bf;
typedef __attribute__((ext_vector_type(8)))  float  v8f;

__device__ __forceinline__ v8f wmma_bf16(v16bf a, v16bf b, v8f c) {
  // D = A(16x32 bf16) * B(32x16 bf16) + C(16x16 f32)
  return __builtin_amdgcn_wmma_f32_16x16x32_bf16(false, a, false, b, (short)0, c,
                                                 false, false);
}

// CDNA5 async global->LDS copy (16 B per lane), tracked by ASYNCcnt.
__device__ __forceinline__ void async_b128_to_lds(const __bf16* gptr, __bf16* lptr) {
  unsigned lds_off = (unsigned)(uintptr_t)lptr;          // flat-LDS aperture: low 32 bits
  unsigned long long ga = (unsigned long long)(uintptr_t)gptr;
  asm volatile("global_load_async_to_lds_b128 %0, %1, off"
               :: "v"(lds_off), "v"(ga) : "memory");
}
__device__ __forceinline__ void wait_async0() {
  asm volatile("s_wait_asynccnt 0x0" ::: "memory");
}

// ---------------------------------------------------------------------------
// f32 -> bf16 conversion (weights)
// ---------------------------------------------------------------------------
__global__ void f2bf_kernel(const float* __restrict__ in, __bf16* __restrict__ out, int n) {
  int i = blockIdx.x * 256 + threadIdx.x;
  if (i < n) out[i] = (__bf16)in[i];
}

// ---------------------------------------------------------------------------
// Block-wide sum reduction of NV values per thread (256 threads, 8 waves)
// ---------------------------------------------------------------------------
template <int NV>
__device__ __forceinline__ void blockReduceSumN(float* v, float* red /* NV*8 */) {
  const int lane = threadIdx.x & 31;
  const int w    = threadIdx.x >> 5;
  __syncthreads();
  for (int i = 0; i < NV; ++i) {
    float x = v[i];
    for (int o = 16; o >= 1; o >>= 1) x += __shfl_down(x, o, 32);
    if (lane == 0) red[i * 8 + w] = x;
  }
  __syncthreads();
  if (threadIdx.x < NV) {
    float s = 0.f;
    for (int k = 0; k < 8; ++k) s += red[threadIdx.x * 8 + k];
    red[threadIdx.x * 8] = s;
  }
  __syncthreads();
  for (int i = 0; i < NV; ++i) v[i] = red[i * 8];
}

// ---------------------------------------------------------------------------
// Depth-softmax attention residual over layer stack, fused with LayerNorm.
// One block per (b,t). NL = 6 (sources only) or 7 (sources + extra layer).
// ---------------------------------------------------------------------------
template <int NL>
__global__ __launch_bounds__(256) void attnres_ln_kernel(
    const float* __restrict__ src,   // [LL, B*T, D]
    const float* __restrict__ extra, // [B*T, D] or nullptr
    const float* __restrict__ proj, const float* __restrict__ rmsw,
    const float* __restrict__ lnw, const float* __restrict__ lnb,
    __bf16* __restrict__ out)        // [B*T, D]
{
  const int bt  = blockIdx.x;
  const int tid = threadIdx.x;
  __shared__ float red[2 * NL * 8];

  float v[NL][4];
  float part[2 * NL];
  for (int l = 0; l < NL; ++l) {
    const float* p = (l < LL) ? (src + ((size_t)l * BT + bt) * DD)
                              : (extra + (size_t)bt * DD);
    float ss = 0.f, dp = 0.f;
    for (int k = 0; k < 4; ++k) {
      int d = tid + k * 256;
      float x = p[d];
      v[l][k] = x;
      ss += x * x;
      dp += proj[d] * rmsw[d] * x;
    }
    part[2 * l] = ss;
    part[2 * l + 1] = dp;
  }
  blockReduceSumN<2 * NL>(part, red);

  // softmax over layer axis
  float logit[NL], mx = -3.0e38f;
  for (int l = 0; l < NL; ++l) {
    float inv = rsqrtf(part[2 * l] * (1.f / DD) + RMS_EPS);
    logit[l] = part[2 * l + 1] * inv;
    mx = fmaxf(mx, logit[l]);
  }
  float s = 0.f;
  for (int l = 0; l < NL; ++l) { logit[l] = __expf(logit[l] - mx); s += logit[l]; }
  float invs = 1.f / s;

  float h[4];
  for (int k = 0; k < 4; ++k) {
    float acc = 0.f;
    for (int l = 0; l < NL; ++l) acc += logit[l] * invs * v[l][k];
    h[k] = acc;
  }

  // LayerNorm
  float st[2] = {0.f, 0.f};
  for (int k = 0; k < 4; ++k) { st[0] += h[k]; st[1] += h[k] * h[k]; }
  blockReduceSumN<2>(st, red);
  float mu   = st[0] * (1.f / DD);
  float var  = st[1] * (1.f / DD) - mu * mu;
  float rinv = rsqrtf(var + LN_EPS);
  for (int k = 0; k < 4; ++k) {
    int d = tid + k * 256;
    out[(size_t)bt * DD + d] = (__bf16)((h[k] - mu) * rinv * lnw[d] + lnb[d]);
  }
}

// ---------------------------------------------------------------------------
// WMMA bf16 GEMM: C[M,N] = A[M,K] @ B[N,K]^T, fused epilogues.
// 256 threads = 8 waves; block tile 128x128; each wave 64x32 (4x2 wmma tiles).
// Double-buffered 128x64 LDS panels filled with async global->LDS copies.
// ---------------------------------------------------------------------------
enum { EPI_QKV = 0, EPI_OUTPROJ = 1, EPI_GELU = 2, EPI_FFN2 = 3 };

#define LSTR 80  // LDS row stride in bf16 (160 B, 32 B-aligned fragments)

__device__ __forceinline__ void stage_tile_async(const __bf16* gbase, int K, int k0,
                                                 __bf16* lds, int tid) {
  // 128 rows x 64 cols bf16; each thread issues 4x 16-byte async copies.
  for (int j = 0; j < 4; ++j) {
    int ch = tid * 4 + j;       // 0..1023
    int r  = ch >> 3;           // 0..127
    int c  = (ch & 7) * 8;      // 0..56
    async_b128_to_lds(gbase + (size_t)r * K + k0 + c, lds + r * LSTR + c);
  }
}

template <int EPI>
__global__ __launch_bounds__(256) void gemm_kernel(
    const __bf16* __restrict__ A, const __bf16* __restrict__ Bw,
    const float* __restrict__ bias, const float* __restrict__ res,
    float* __restrict__ outf, __bf16* __restrict__ outb,
    __bf16* __restrict__ outq, __bf16* __restrict__ outk, __bf16* __restrict__ outv,
    int N, int K)
{
  __shared__ alignas(64) __bf16 As[2][128 * LSTR];
  __shared__ alignas(64) __bf16 Bs[2][128 * LSTR];
  const int tid  = threadIdx.x;
  const int lane = tid & 31;
  const int wid  = tid >> 5;
  const int wm   = wid >> 2;   // 0..1  (64-row slab)
  const int wn   = wid & 3;    // 0..3  (32-col slab)
  const int bm   = blockIdx.y * 128;
  const int bn   = blockIdx.x * 128;
  const int lr   = lane & 15;
  const int lh   = lane >> 4;

  const __bf16* Ag = A + (size_t)bm * K;
  const __bf16* Bg = Bw + (size_t)bn * K;

  v8f acc[4][2] = {};

  int cur = 0;
  stage_tile_async(Ag, K, 0, As[0], tid);
  stage_tile_async(Bg, K, 0, Bs[0], tid);

  for (int k0 = 0; k0 < K; k0 += 64) {
    wait_async0();        // this wave's async copies into `cur` complete
    __syncthreads();      // workgroup-wide: `cur` ready, `cur^1` fully consumed
    if (k0 + 64 < K) {
      stage_tile_async(Ag, K, k0 + 64, As[cur ^ 1], tid);
      stage_tile_async(Bg, K, k0 + 64, Bs[cur ^ 1], tid);
    }
    for (int ks = 0; ks < 64; ks += 32) {
      v16bf bf[2];
      for (int j = 0; j < 2; ++j)
        bf[j] = *(const v16bf*)&Bs[cur][(wn * 32 + j * 16 + lr) * LSTR + ks + lh * 16];
      for (int i = 0; i < 4; ++i) {
        v16bf af = *(const v16bf*)&As[cur][(wm * 64 + i * 16 + lr) * LSTR + ks + lh * 16];
        for (int j = 0; j < 2; ++j)
          acc[i][j] = wmma_bf16(af, bf[j], acc[i][j]);
      }
    }
    cur ^= 1;
  }

  // Epilogue. C layout: VGPR e -> M = e + 8*(lane>>4); N = lane&15.
  for (int i = 0; i < 4; ++i) {
    for (int j = 0; j < 2; ++j) {
      for (int e = 0; e < 8; ++e) {
        int row = bm + wm * 64 + i * 16 + e + (lh << 3);
        int col = bn + wn * 32 + j * 16 + lr;
        float val = acc[i][j][e];
        if (EPI == EPI_QKV) {
          val += bias[col];
          int sect = col >> 10, c2 = col & 1023;
          int h = c2 >> 6, hd = c2 & 63;
          int b = row >> 10, t = row & 1023;
          size_t qi = (((size_t)(b * HH + h)) * TT + t) * HDIM + hd;
          if (sect == 0)      outq[qi] = (__bf16)(val * 0.125f);  // 1/sqrt(hd)
          else if (sect == 1) outk[qi] = (__bf16)val;
          else                outv[(((size_t)(b * HH + h)) * HDIM + hd) * TT + t] = (__bf16)val;
        } else if (EPI == EPI_OUTPROJ) {
          outf[(size_t)row * DD + col] = val + bias[col] + res[(size_t)row * DD + col];
        } else if (EPI == EPI_GELU) {
          float x = val + bias[col];
          outb[(size_t)row * N + col] =
              (__bf16)(0.5f * x * (1.f + erff(x * 0.70710678118f)));
        } else { // EPI_FFN2
          outf[(size_t)row * DD + col] = val + bias[col] + res[(size_t)row * DD + col];
        }
      }
    }
  }
}

// ---------------------------------------------------------------------------
// Causal multi-head attention. One block per (b, h, 16-row q-tile).
// q,k: [B,H,T,hd] bf16 (q pre-scaled). vT: [B,H,hd,T] bf16.
// ---------------------------------------------------------------------------
__global__ __launch_bounds__(256) void attn_kernel(
    const __bf16* __restrict__ q, const __bf16* __restrict__ k,
    const __bf16* __restrict__ vT, __bf16* __restrict__ ctx)
{
  __shared__ alignas(64) float  sc[16][1024];  // score strip (f32)
  __shared__ alignas(64) __bf16 pb[16][1056];  // softmax probs (bf16)
  const int qt  = blockIdx.x;
  const int h   = blockIdx.y;
  const int b   = blockIdx.z;
  const int tid = threadIdx.x;
  const int lane = tid & 31, wid = tid >> 5;
  const int lr = lane & 15, lh = lane >> 4;
  const int nkt    = qt + 1;
  const int kend32 = (nkt * 16 + 31) & ~31;

  // Phase 1: S = q_tile @ k^T (WMMA), tiles distributed over 8 waves.
  const __bf16* qbase = q + (((size_t)(b * HH + h)) * TT + qt * 16) * HDIM;
  v16bf a0 = *(const v16bf*)&qbase[lr * HDIM + lh * 16];
  v16bf a1 = *(const v16bf*)&qbase[lr * HDIM + lh * 16 + 32];
  const __bf16* kbase = k + ((size_t)(b * HH + h)) * TT * HDIM;
  for (int kt = wid; kt < nkt; kt += 8) {
    v16bf b0 = *(const v16bf*)&kbase[(size_t)(kt * 16 + lr) * HDIM + lh * 16];
    v16bf b1 = *(const v16bf*)&kbase[(size_t)(kt * 16 + lr) * HDIM + lh * 16 + 32];
    v8f c = {};
    c = wmma_bf16(a0, b0, c);
    c = wmma_bf16(a1, b1, c);
    for (int e = 0; e < 8; ++e)
      sc[e + (lh << 3)][kt * 16 + lr] = c[e];
  }
  __syncthreads();

  // Phase 2: causal softmax, 16 threads per row.
  {
    const int g  = tid >> 4;          // row in tile
    const int li = tid & 15;
    const int qg = qt * 16 + g;       // mask: col <= qg
    float mx = -3.0e38f;
    for (int c = li; c <= qg; c += 16) mx = fmaxf(mx, sc[g][c]);
    for (int o = 8; o >= 1; o >>= 1) mx = fmaxf(mx, __shfl_xor(mx, o, 16));
    float s = 0.f;
    for (int c = li; c <= qg; c += 16) {
      float e = __expf(sc[g][c] - mx);
      sc[g][c] = e;
      s += e;
    }
    for (int o = 8; o >= 1; o >>= 1) s += __shfl_xor(s, o, 16);
    float inv = 1.f / s;
    for (int c = li; c < kend32; c += 16)
      pb[g][c] = (c <= qg) ? (__bf16)(sc[g][c] * inv) : (__bf16)0.f;
  }
  __syncthreads();

  // Phase 3: ctx = P @ V (WMMA), one 16-col hd tile per wave (waves 0..3).
  if (wid < 4) {
    const int nt = wid;
    const __bf16* vbase = vT + ((size_t)(b * HH + h)) * HDIM * TT;
    v8f c = {};
    for (int k0 = 0; k0 < kend32; k0 += 32) {
      v16bf af = *(const v16bf*)&pb[lr][k0 + lh * 16];
      v16bf bf = *(const v16bf*)&vbase[(size_t)(nt * 16 + lr) * TT + k0 + lh * 16];
      c = wmma_bf16(af, bf, c);
    }
    for (int e = 0; e < 8; ++e) {
      int row = e + (lh << 3);
      int col = nt * 16 + lr;
      ctx[((size_t)b * TT + qt * 16 + row) * DD + h * HDIM + col] = (__bf16)c[e];
    }
  }
}

// ---------------------------------------------------------------------------
// Host-side orchestration
// ---------------------------------------------------------------------------
extern "C" void kernel_launch(void* const* d_in, const int* in_sizes, int n_in,
                              void* d_out, int out_size, void* d_ws, size_t ws_size,
                              hipStream_t stream) {
  (void)in_sizes; (void)n_in; (void)out_size; (void)ws_size;
  const float* sources    = (const float*)d_in[0];
  const float* w_attn     = (const float*)d_in[1];
  const float* rms_attn_w = (const float*)d_in[2];
  const float* w_ffn      = (const float*)d_in[3];
  const float* rms_ffn_w  = (const float*)d_in[4];
  const float* ln_attn_w  = (const float*)d_in[5];
  const float* ln_attn_b  = (const float*)d_in[6];
  const float* ln_ffn_w   = (const float*)d_in[7];
  const float* ln_ffn_b   = (const float*)d_in[8];
  const float* in_proj_w  = (const float*)d_in[9];
  const float* in_proj_b  = (const float*)d_in[10];
  const float* out_proj_w = (const float*)d_in[11];
  const float* out_proj_b = (const float*)d_in[12];
  const float* ffn_w1     = (const float*)d_in[13];
  const float* ffn_b1     = (const float*)d_in[14];
  const float* ffn_w2     = (const float*)d_in[15];
  const float* ffn_b2     = (const float*)d_in[16];
  float* out = (float*)d_out;

  char* ws = (char*)d_ws;
  size_t off = 0;
  auto alloc = [&](size_t bytes) {
    size_t r = off;
    off += (bytes + 255) & ~(size_t)255;
    return r;
  };
  __bf16* wq  = (__bf16*)(ws + alloc((size_t)3 * DD * DD * 2));
  __bf16* wo  = (__bf16*)(ws + alloc((size_t)DD * DD * 2));
  __bf16* w1  = (__bf16*)(ws + alloc((size_t)DFF * DD * 2));
  __bf16* w2  = (__bf16*)(ws + alloc((size_t)DD * DFF * 2));
  __bf16* xln = (__bf16*)(ws + alloc((size_t)BT * DD * 2));
  __bf16* qb  = (__bf16*)(ws + alloc((size_t)BT * DD * 2));
  __bf16* kb  = (__bf16*)(ws + alloc((size_t)BT * DD * 2));
  __bf16* vb  = (__bf16*)(ws + alloc((size_t)BT * DD * 2));  // transposed [B,H,hd,T]
  __bf16* ctx = (__bf16*)(ws + alloc((size_t)BT * DD * 2));
  float*  vla = (float*)(ws + alloc((size_t)BT * DD * 4));
  __bf16* yln = (__bf16*)(ws + alloc((size_t)BT * DD * 2));
  __bf16* act = (__bf16*)(ws + alloc((size_t)BT * DFF * 2));

  // 1) weight conversion to bf16
  f2bf_kernel<<<(3 * DD * DD + 255) / 256, 256, 0, stream>>>(in_proj_w, wq, 3 * DD * DD);
  f2bf_kernel<<<(DD * DD + 255) / 256, 256, 0, stream>>>(out_proj_w, wo, DD * DD);
  f2bf_kernel<<<(DFF * DD + 255) / 256, 256, 0, stream>>>(ffn_w1, w1, DFF * DD);
  f2bf_kernel<<<(DD * DFF + 255) / 256, 256, 0, stream>>>(ffn_w2, w2, DD * DFF);

  // 2) depth-softmax residual #1 + LN -> x_ln (bf16)
  attnres_ln_kernel<6><<<BT, 256, 0, stream>>>(sources, nullptr, w_attn, rms_attn_w,
                                               ln_attn_w, ln_attn_b, xln);
  // 3) qkv projection; scatter into q/k/vT
  gemm_kernel<EPI_QKV><<<dim3(3 * DD / 128, BT / 128), 256, 0, stream>>>(
      xln, wq, in_proj_b, nullptr, nullptr, nullptr, qb, kb, vb, 3 * DD, DD);
  // 4) causal attention
  attn_kernel<<<dim3(TT / 16, HH, BB), 256, 0, stream>>>(qb, kb, vb, ctx);
  // 5) out_proj + bias + residual (v_prev = sources[5]) -> vla (f32)
  gemm_kernel<EPI_OUTPROJ><<<dim3(DD / 128, BT / 128), 256, 0, stream>>>(
      ctx, wo, out_proj_b, sources + (size_t)5 * BT * DD, vla, nullptr,
      nullptr, nullptr, nullptr, DD, DD);
  // 6) depth-softmax residual #2 (7 layers) + LN -> y_ln (bf16)
  attnres_ln_kernel<7><<<BT, 256, 0, stream>>>(sources, vla, w_ffn, rms_ffn_w,
                                               ln_ffn_w, ln_ffn_b, yln);
  // 7) FFN1 + exact GELU -> act (bf16)
  gemm_kernel<EPI_GELU><<<dim3(DFF / 128, BT / 128), 256, 0, stream>>>(
      yln, w1, ffn_b1, nullptr, nullptr, act, nullptr, nullptr, nullptr, DFF, DD);
  // 8) FFN2 + bias + vla residual -> final output (f32)
  gemm_kernel<EPI_FFN2><<<dim3(DD / 128, BT / 128), 256, 0, stream>>>(
      act, w2, ffn_b2, vla, out, nullptr, nullptr, nullptr, nullptr, DD, DFF);
}
